// StaticModelBatched_59871844106303
// MI455X (gfx1250) — matrix operand order repaired and checked by reading, not compiled, not run on hardware
//
#include <hip/hip_runtime.h>
#include <hip/hip_bf16.h>

typedef __attribute__((ext_vector_type(16))) __bf16 v16bf;
typedef __attribute__((ext_vector_type(8)))  float  v8f;

#define DFEAT 64
#define GN_EPS 1e-5f

// ---------------------------------------------------------------------------
// Utility fills
// ---------------------------------------------------------------------------
__global__ void fill_f32(float* __restrict__ p, float v, long long n) {
    long long i = (long long)blockIdx.x * blockDim.x + threadIdx.x;
    if (i < n) p[i] = v;
}

// ---------------------------------------------------------------------------
// gcn_norm: degree accumulation + symmetric normalization (incl. self loops)
// ---------------------------------------------------------------------------
__global__ void deg_accum(const int* __restrict__ ei, const float* __restrict__ ew,
                          float* __restrict__ deg, int E) {
    int e = blockIdx.x * blockDim.x + threadIdx.x;
    if (e < E) atomicAdd(&deg[ei[E + e]], ew[e]);   // col index
}

__global__ void compute_norm(const int* __restrict__ ei, const float* __restrict__ ew,
                             const float* __restrict__ deg, float* __restrict__ nrm,
                             int E, int N) {
    int e = blockIdx.x * blockDim.x + threadIdx.x;
    if (e >= E + N) return;
    if (e < E) {
        int r = ei[e], c = ei[E + e];
        float dr = deg[r], dc = deg[c];
        float ir = dr > 0.f ? rsqrtf(dr) : 0.f;
        float ic = dc > 0.f ? rsqrtf(dc) : 0.f;
        nrm[e] = ir * ew[e] * ic;
    } else {
        int i = e - E;
        float d = deg[i];
        float di = d > 0.f ? rsqrtf(d) : 0.f;
        nrm[e] = di * di;            // self loop, weight 1
    }
}

// ---------------------------------------------------------------------------
// WMMA GEMM: Y[nrows x NCOL] = X[nrows x 64] * W[64 x NCOL] (+ bias)
// One 16x16 C tile per wave; K=64 -> two chained v_wmma_f32_16x16x32_bf16.
// Lane layouts per CDNA5 ISA 7.12.2 (wave32).
// ---------------------------------------------------------------------------
template <int NCOL, int RT>
__global__ __launch_bounds__(256)
void gemm_wmma(const float* __restrict__ X, const float* __restrict__ W,
               const float* __restrict__ bias, float* __restrict__ Y, int nrows) {
    constexpr int CT = NCOL / 16;               // col tiles
    const int wave = threadIdx.x >> 5;
    const int lane = threadIdx.x & 31;
    const int rt   = wave / CT;
    const int ct   = wave % CT;
    const int m0   = (blockIdx.x * RT + rt) * 16;
    const int n0   = ct * 16;
    if (m0 >= nrows) return;                    // wave-uniform: EXEC all-1s inside
    const int half = lane >> 4;
    const int mr   = lane & 15;                 // A row / B,C column

    // A: 16x32 bf16 per WMMA; lane holds row m=mr; VGPR j -> K = g*16 + half*8 + jj*2
    const float* xrow = X + (size_t)(m0 + mr) * DFEAT;
    v16bf a0, a1;
#pragma unroll
    for (int j = 0; j < 8; ++j) {
        int g = j >> 2, jj = j & 3;
        int k = g * 16 + half * 8 + jj * 2;
        a0[2 * j]     = (__bf16)xrow[k];
        a0[2 * j + 1] = (__bf16)xrow[k + 1];
        a1[2 * j]     = (__bf16)xrow[32 + k];
        a1[2 * j + 1] = (__bf16)xrow[32 + k + 1];
    }
    // B: 32x16 bf16; lane holds col n=mr; VGPR j -> K = (j>>2)*16 + half*8 + (j&3)*2 (matches A)
    v16bf b0, b1;
#pragma unroll
    for (int j = 0; j < 8; ++j) {
        int k = (j >> 2) * 16 + half * 8 + (j & 3) * 2;
        b0[2 * j]     = (__bf16)W[(size_t)k * NCOL + n0 + mr];
        b0[2 * j + 1] = (__bf16)W[(size_t)(k + 1) * NCOL + n0 + mr];
        b1[2 * j]     = (__bf16)W[(size_t)(32 + k) * NCOL + n0 + mr];
        b1[2 * j + 1] = (__bf16)W[(size_t)(33 + k) * NCOL + n0 + mr];
    }
    v8f c = {};
    c = __builtin_amdgcn_wmma_f32_16x16x32_bf16(false, a0, false, b0, (short)0, c, false, false);
    c = __builtin_amdgcn_wmma_f32_16x16x32_bf16(false, a1, false, b1, (short)0, c, false, false);

    float bv = bias ? bias[n0 + mr] : 0.0f;
    // C: VGPR v -> row M = v + 8*half, col N = mr
#pragma unroll
    for (int v = 0; v < 8; ++v) {
        int m = m0 + v + 8 * half;
        Y[(size_t)m * NCOL + n0 + mr] = c[v] + bv;
    }
}

// ---------------------------------------------------------------------------
// Scatter-add of normalized messages: acc[col] += norm[e] * xw[row]
// 64 consecutive lanes per edge -> coalesced gather and contiguous L2 atomics.
// ---------------------------------------------------------------------------
__global__ void scatter_add(const float* __restrict__ xw, const int* __restrict__ ei,
                            const float* __restrict__ nrm, float* __restrict__ acc,
                            int E, int N) {
    long long t = (long long)blockIdx.x * blockDim.x + threadIdx.x;
    long long total = (long long)(E + N) * DFEAT;
    if (t >= total) return;
    int d = (int)(t & (DFEAT - 1));
    int e = (int)(t >> 6);
    int r, c;
    if (e < E) { r = ei[e]; c = ei[E + e]; } else { r = c = e - E; }
    float v = nrm[e] * xw[(size_t)r * DFEAT + d];
    atomicAdd(&acc[(size_t)c * DFEAT + d], v);
}

// ---------------------------------------------------------------------------
// GraphNorm single-pass stats: per (graph,feature) sum and sum-of-squares of
// (acc + bias). grid = (B, nchunks), block = 64 threads (one per feature).
// ---------------------------------------------------------------------------
__global__ void graph_stats(const float* __restrict__ acc, const float* __restrict__ bias,
                            float* __restrict__ stats, int per_graph, int nchunks, int B) {
    int b  = blockIdx.x;
    int ch = blockIdx.y;
    int d  = threadIdx.x;
    int chunk = (per_graph + nchunks - 1) / nchunks;
    int n0 = b * per_graph + ch * chunk;
    int n1 = min(b * per_graph + per_graph, n0 + chunk);
    float bv = bias[d];
    float s1 = 0.f, s2 = 0.f;
    for (int n = n0; n < n1; ++n) {
        float v = acc[(size_t)n * DFEAT + d] + bv;
        s1 += v;
        s2 += v * v;
    }
    atomicAdd(&stats[b * DFEAT + d], s1);
    atomicAdd(&stats[(size_t)B * DFEAT + b * DFEAT + d], s2);
}

// ---------------------------------------------------------------------------
// GraphNorm apply (+bias fold, + optional ReLU), in-place capable.
// var = E[x^2] - 2*ms*m*E[x] + (ms*m)^2  with m = E[x]
// ---------------------------------------------------------------------------
__global__ void graph_norm_apply(const float* __restrict__ acc, const float* __restrict__ bias,
                                 const float* __restrict__ stats, const int* __restrict__ batch,
                                 const float* __restrict__ gw, const float* __restrict__ gb,
                                 const float* __restrict__ gms, float* __restrict__ out,
                                 int N, int B, float inv_cnt, int relu) {
    long long t = (long long)blockIdx.x * blockDim.x + threadIdx.x;
    if (t >= (long long)N * DFEAT) return;
    int d = (int)(t & (DFEAT - 1));
    int n = (int)(t >> 6);
    int b = batch[n];
    float S1 = stats[b * DFEAT + d];
    float S2 = stats[(size_t)B * DFEAT + b * DFEAT + d];
    float m  = S1 * inv_cnt;
    float mm = gms[d] * m;
    float var = S2 * inv_cnt - 2.0f * mm * m + mm * mm;
    float v = acc[t] + bias[d];
    float y = gw[d] * (v - mm) * rsqrtf(var + GN_EPS) + gb[d];
    if (relu) y = fmaxf(y, 0.0f);
    out[t] = y;
}

// ---------------------------------------------------------------------------
extern "C" void kernel_launch(void* const* d_in, const int* in_sizes, int n_in,
                              void* d_out, int out_size, void* d_ws, size_t ws_size,
                              hipStream_t stream) {
    (void)n_in; (void)out_size; (void)ws_size;
    const float* x_in  = (const float*)d_in[0];
    const int*   ei    = (const int*)d_in[1];
    const float* ew    = (const float*)d_in[2];
    const int*   batch = (const int*)d_in[3];
    const float* Ws    = (const float*)d_in[4];
    const float* bs    = (const float*)d_in[5];
    const float* gn_w  = (const float*)d_in[6];
    const float* gn_b  = (const float*)d_in[7];
    const float* gn_ms = (const float*)d_in[8];
    const float* Wr    = (const float*)d_in[9];
    const float* br    = (const float*)d_in[10];
    float* out = (float*)d_out;

    const int N = in_sizes[0] / DFEAT;   // 80000
    const int E = in_sizes[2];           // 1280000
    const int B = 16;
    const int per_graph = N / B;

    float* ws    = (float*)d_ws;
    float* xw    = ws;                               // N*64
    float* acc   = xw  + (size_t)N * DFEAT;          // N*64
    float* nrm   = acc + (size_t)N * DFEAT;          // E+N
    float* deg   = nrm + (size_t)(E + N);            // N
    float* stats = deg + N;                          // 2*B*64

    // gcn_norm
    fill_f32<<<(N + 255) / 256, 256, 0, stream>>>(deg, 1.0f, N);
    deg_accum<<<(E + 255) / 256, 256, 0, stream>>>(ei, ew, deg, E);
    compute_norm<<<(E + N + 255) / 256, 256, 0, stream>>>(ei, ew, deg, nrm, E, N);

    const long long nd = (long long)N * DFEAT;
    const float inv_cnt = 1.0f / (float)per_graph;

    const float* src = x_in;
    for (int i = 0; i < 5; ++i) {
        // xw = src @ Ws[i]   (WMMA bf16 -> f32)
        gemm_wmma<DFEAT, 2><<<N / 32, 256, 0, stream>>>(src, Ws + (size_t)i * DFEAT * DFEAT,
                                                        nullptr, xw, N);
        // zero accumulator + stats (stream-ordered after gemm consumed src==acc)
        fill_f32<<<(int)((nd + 255) / 256), 256, 0, stream>>>(acc, 0.0f, nd);
        fill_f32<<<(2 * B * DFEAT + 255) / 256, 256, 0, stream>>>(stats, 0.0f, 2 * B * DFEAT);
        // message passing scatter
        long long tot = (long long)(E + N) * DFEAT;
        scatter_add<<<(int)((tot + 255) / 256), 256, 0, stream>>>(xw, ei, nrm, acc, E, N);
        // GraphNorm stats + apply (bias folded), ReLU on layers 0..3
        dim3 sg(B, 32);
        graph_stats<<<sg, DFEAT, 0, stream>>>(acc, bs + (size_t)i * DFEAT, stats, per_graph, 32, B);
        graph_norm_apply<<<(int)((nd + 255) / 256), 256, 0, stream>>>(
            acc, bs + (size_t)i * DFEAT, stats, batch,
            gn_w + (size_t)i * DFEAT, gn_b + (size_t)i * DFEAT, gn_ms + (size_t)i * DFEAT,
            acc, N, B, inv_cnt, (i < 4) ? 1 : 0);
        src = acc;
    }
    // readout: out = src @ Wr + br   (N x 64 @ 64 x 32)
    gemm_wmma<32, 4><<<N / 64, 256, 0, stream>>>(src, Wr, br, out, N);
}
